// FFB_encoder_51780125720905
// MI455X (gfx1250) — matrix operand order repaired and checked by reading, not compiled, not run on hardware
//
#include <hip/hip_runtime.h>
#include <hip/hip_bf16.h>
#include <stdint.h>

typedef __attribute__((ext_vector_type(16))) __bf16 v16bf;
typedef __attribute__((ext_vector_type(8)))  float  v8f;

#define INV2PI 0.15915494309189535f

__device__ __forceinline__ unsigned short f2bf(float f) {
    unsigned u = __builtin_bit_cast(unsigned, f);
    u += 0x7FFFu + ((u >> 16) & 1u);           // round-to-nearest-even
    return (unsigned short)(u >> 16);
}
__device__ __forceinline__ float bf2f(unsigned short s) {
    unsigned u = ((unsigned)s) << 16;
    return __builtin_bit_cast(float, u);
}

union Frag { uint4 q[2]; v16bf v; };

// A-fragment (16x32 bf16, MxK): lane m=L%16, h=L/16.
// VGPR0-3: K = kk*32 + h*8 + [0..7]; VGPR4-7: K = kk*32 + 16 + h*8 + [0..7].
__device__ __forceinline__ v16bf load_a(const unsigned short* row, int kk, int h) {
    Frag f;
    f.q[0] = *(const uint4*)(row + kk * 32 + h * 8);
    f.q[1] = *(const uint4*)(row + kk * 32 + 16 + h * 8);
    return f.v;
}
// B-fragment (32x16 bf16, KxN) from transposed [N][K] storage: lane n=L%16, h=L/16.
// VGPR0-3: K = kk*32 + h*8 + [0..7]; VGPR4-7: K = kk*32 + 16 + h*8 + [0..7].
__device__ __forceinline__ v16bf load_b(const unsigned short* nrow, int kk, int h) {
    Frag f;
    f.q[0] = *(const uint4*)(nrow + kk * 32 + h * 8);
    f.q[1] = *(const uint4*)(nrow + kk * 32 + 16 + h * 8);
    return f.v;
}

__device__ __forceinline__ v8f wmma_bf16(v16bf a, v16bf b, v8f c) {
    return __builtin_amdgcn_wmma_f32_16x16x32_bf16(false, a, false, b, (short)0, c,
                                                   false, false);
}

// Async global->LDS DMA (CDNA5): 16B per lane, tracked by ASYNCcnt.
// ldsaddr: LDS byte address (low 32 bits of generic shared pointer);
// goff:    per-lane 32-bit byte offset added to uniform 64-bit base (GVS mode).
__device__ __forceinline__ void async_cp16(unsigned ldsaddr, unsigned goff,
                                           const void* sbase) {
    asm volatile("global_load_async_to_lds_b128 %0, %1, %2"
                 :: "v"(ldsaddr), "v"(goff), "s"(sbase)
                 : "memory");
}
__device__ __forceinline__ void wait_async0() {
    asm volatile("s_wait_asynccnt 0x0" ::: "memory");
}

// ---------------------------------------------------------------------------
// Prep: transpose + convert weights to bf16 in workspace.
//   apad [4][256][32] : apad[l][n][k] = ffn_A[l][k][n] * 2^l  (k<8), else 0
//   wst  [4][256][256]: wst[l][n][k]  = Ws[l][k][n]
//   wht  [4][256][256]: wht[l][n][k]  = Wh[l][k][n]
// ---------------------------------------------------------------------------
__global__ void ffb_prep(const float* __restrict__ ffn_A,
                         const float* __restrict__ Ws,
                         const float* __restrict__ Wh,
                         unsigned short* __restrict__ apad,
                         unsigned short* __restrict__ wst,
                         unsigned short* __restrict__ wht) {
    int i = blockIdx.x * blockDim.x + threadIdx.x;
    if (i < 4 * 256 * 32) {
        int k = i & 31, n = (i >> 5) & 255, l = i >> 13;
        float v = 0.f;
        if (k < 8) {
            float sigma = (float)(1 << l);   // BASE_SIGMA * EXP_SIGMA^l = 2^l
            v = ffn_A[((size_t)l * 8 + k) * 256 + n] * sigma;
        }
        apad[i] = f2bf(v);
    }
    for (int j = i; j < 4 * 256 * 256; j += gridDim.x * blockDim.x) {
        int k = j & 255, n = (j >> 8) & 255, l = j >> 16;
        size_t src = ((size_t)l * 256 + k) * 256 + n;
        wst[j] = f2bf(Ws[src]);
        wht[j] = f2bf(Wh[src]);
    }
}

// ---------------------------------------------------------------------------
// Fused FFB encoder. 1 block = 64 points, 256 threads = 8 waves (wave32).
// Wave w: mtile = w>>1 (16 rows), nhalf = w&1 (128 cols -> 8 N-tiles).
// ---------------------------------------------------------------------------
#define XSTR 264   // padded row stride (bf16 elems): 528B = 33*16B, shifts banks

__global__ __launch_bounds__(256, 1)
void ffb_main(const float* __restrict__ in_pos,
              const float* __restrict__ table,
              const float* __restrict__ W0,
              const float* __restrict__ b0,
              const float* __restrict__ bs,
              const float* __restrict__ bh,
              const unsigned short* __restrict__ apad,
              const unsigned short* __restrict__ wst,
              const unsigned short* __restrict__ wht,
              float* __restrict__ out) {
    extern __shared__ char smem[];
    unsigned short* xs = (unsigned short*)smem;      // [64][XSTR]  activations (bf16)
    unsigned short* wl = xs + 64 * XSTR;             // [256][XSTR] staged weight (bf16)
    unsigned short* gx = wl + 256 * XSTR;            // [64][XSTR]  grid-branch sin (bf16)
    unsigned short* gp = gx + 64 * XSTR;             // [64][32]    grid feats, K-padded
    float* posb = (float*)(gp + 64 * 32);            // [64][3]     staged positions

    const int tid   = threadIdx.x;
    const int lane  = tid & 31;
    const int wave  = tid >> 5;
    const int mtile = wave >> 1;
    const int nhalf = wave & 1;
    const int h     = lane >> 4;
    const int ln    = lane & 15;
    const int tileBase = blockIdx.x * 64;

    // stage positions + zero K-padding of grid features
    if (tid < 192) posb[tid] = in_pos[(size_t)tileBase * 3 + tid];
    for (int i = tid; i < 64 * 32; i += 256) gp[i] = 0;
    __syncthreads();

    // ---- layer 0: x = sin(pos @ W0 + b0); thread owns column n = tid ----
    {
        const int n = tid;
        float w0x = W0[n], w0y = W0[256 + n], w0z = W0[512 + n];
        float bb = b0[n];
        for (int m = 0; m < 64; ++m) {
            float v = fmaf(posb[m * 3 + 0], w0x,
                      fmaf(posb[m * 3 + 1], w0y,
                      fmaf(posb[m * 3 + 2], w0z, bb)));
            xs[m * XSTR + n] = f2bf(__builtin_amdgcn_sinf(v * INV2PI));
        }
    }

    v8f xout[8];
#pragma unroll
    for (int i = 0; i < 8; ++i)
        xout[i] = (v8f){0.f, 0.f, 0.f, 0.f, 0.f, 0.f, 0.f, 0.f};

    const unsigned short* xrow = xs + (mtile * 16 + ln) * XSTR;  // A-frag row (m = ln)
    const unsigned short* grow = gp + (mtile * 16 + ln) * 32;
    const unsigned wl_lds = (unsigned)(uintptr_t)(wl + tid * XSTR);

    for (int l = 0; l < 4; ++l) {
        // ---- issue async stage of Ws_t[l] -> wl (thread copies row n = tid) ----
        {
            const void* src = (const void*)(wst + ((size_t)l << 16));
            unsigned gofs = (unsigned)tid * 512u;   // row byte offset in [256][256] bf16
#pragma unroll
            for (int j = 0; j < 32; ++j)
                async_cp16(wl_lds + j * 16, gofs + j * 16, src);
        }
        // ---- hash-grid encode (overlaps the async copy): one point/thread ----
        if (tid < 64) {
            float resf = (float)(16 << l);
            float px = (posb[tid * 3 + 0] + 1.f) * 0.5f * resf;
            float py = (posb[tid * 3 + 1] + 1.f) * 0.5f * resf;
            float pz = (posb[tid * 3 + 2] + 1.f) * 0.5f * resf;
            float fx = floorf(px), fy = floorf(py), fz = floorf(pz);
            float wx = px - fx, wy = py - fy, wz = pz - fz;
            unsigned ix = (unsigned)fx, iy = (unsigned)fy, iz = (unsigned)fz;
            float g[8] = {0, 0, 0, 0, 0, 0, 0, 0};
            const float* tl = table + ((size_t)l << 15) * 8;
#pragma unroll
            for (int c = 0; c < 8; ++c) {
                unsigned ox = (c >> 2) & 1, oy = (c >> 1) & 1, oz = c & 1;
                unsigned hsh = (ix + ox) ^ ((iy + oy) * 2654435761u) ^
                               ((iz + oz) * 805459861u);
                unsigned idx = hsh & 32767u;
                const float4* f4 = (const float4*)(tl + (size_t)idx * 8);
                float4 fa = f4[0], fb = f4[1];
                float w = (ox ? wx : 1.f - wx) * (oy ? wy : 1.f - wy) *
                          (oz ? wz : 1.f - wz);
                g[0] = fmaf(w, fa.x, g[0]); g[1] = fmaf(w, fa.y, g[1]);
                g[2] = fmaf(w, fa.z, g[2]); g[3] = fmaf(w, fa.w, g[3]);
                g[4] = fmaf(w, fb.x, g[4]); g[5] = fmaf(w, fb.y, g[5]);
                g[6] = fmaf(w, fb.z, g[6]); g[7] = fmaf(w, fb.w, g[7]);
            }
#pragma unroll
            for (int q = 0; q < 8; ++q) gp[tid * 32 + q] = f2bf(g[q]);
        }
        wait_async0();
        __syncthreads();

        // ---- grid branch: gx = sin(2*pi * g @ (sigma*A));  K=32, one step ----
        {
            const unsigned short* apl = apad + (size_t)l * 256 * 32;
            v16bf ga = load_a(grow, 0, h);
#pragma unroll
            for (int nt = 0; nt < 8; ++nt) {
                int n = nhalf * 128 + nt * 16 + ln;
                v16bf gb = load_b(apl + (size_t)n * 32, 0, h);
                v8f z = (v8f){0.f, 0.f, 0.f, 0.f, 0.f, 0.f, 0.f, 0.f};
                z = wmma_bf16(ga, gb, z);
#pragma unroll
                for (int r = 0; r < 8; ++r) {   // v_sin_f32 computes sin(2*pi*x)
                    int m = mtile * 16 + h * 8 + r;
                    gx[m * XSTR + n] = f2bf(__builtin_amdgcn_sinf(z[r]));
                }
            }
        }

        // ---- x @ Ws[l] : M=16 strip, N=128 half, K=256 ----
        v8f acc[8];
#pragma unroll
        for (int i = 0; i < 8; ++i)
            acc[i] = (v8f){0.f, 0.f, 0.f, 0.f, 0.f, 0.f, 0.f, 0.f};
#pragma unroll
        for (int kk = 0; kk < 8; ++kk) {
            v16bf a = load_a(xrow, kk, h);
#pragma unroll
            for (int nt = 0; nt < 8; ++nt) {
                int n = nhalf * 128 + nt * 16 + ln;
                acc[nt] = wmma_bf16(a, load_b(wl + n * XSTR, kk, h), acc[nt]);
            }
        }
        __syncthreads();   // everyone done reading xs & wl

        // ---- issue async stage of Wh_t[l] -> wl (overlaps epilogue below) ----
        {
            const void* src = (const void*)(wht + ((size_t)l << 16));
            unsigned gofs = (unsigned)tid * 512u;
#pragma unroll
            for (int j = 0; j < 32; ++j)
                async_cp16(wl_lds + j * 16, gofs + j * 16, src);
        }
        // ---- x = sin(acc + bs) + gx  (bf16 back to LDS) ----
#pragma unroll
        for (int nt = 0; nt < 8; ++nt) {
            int n = nhalf * 128 + nt * 16 + ln;
            float bsv = bs[l * 256 + n];
#pragma unroll
            for (int r = 0; r < 8; ++r) {
                int m = mtile * 16 + h * 8 + r;
                float xv = __builtin_amdgcn_sinf((acc[nt][r] + bsv) * INV2PI) +
                           bf2f(gx[m * XSTR + n]);
                xs[m * XSTR + n] = f2bf(xv);
            }
        }
        wait_async0();
        __syncthreads();   // xs rewritten, wl = Wh ready

        // ---- x @ Wh[l] ----
#pragma unroll
        for (int i = 0; i < 8; ++i)
            acc[i] = (v8f){0.f, 0.f, 0.f, 0.f, 0.f, 0.f, 0.f, 0.f};
#pragma unroll
        for (int kk = 0; kk < 8; ++kk) {
            v16bf a = load_a(xrow, kk, h);
#pragma unroll
            for (int nt = 0; nt < 8; ++nt) {
                int n = nhalf * 128 + nt * 16 + ln;
                acc[nt] = wmma_bf16(a, load_b(wl + n * XSTR, kk, h), acc[nt]);
            }
        }
        // ---- x_out += sin(acc + bh) ----
#pragma unroll
        for (int nt = 0; nt < 8; ++nt) {
            int n = nhalf * 128 + nt * 16 + ln;
            float bhv = bh[l * 256 + n];
#pragma unroll
            for (int r = 0; r < 8; ++r)
                xout[nt][r] += __builtin_amdgcn_sinf((acc[nt][r] + bhv) * INV2PI);
        }
        __syncthreads();   // before next level's staging / encode
    }

    // ---- write output [N,256] f32 ----
#pragma unroll
    for (int nt = 0; nt < 8; ++nt) {
        int n = nhalf * 128 + nt * 16 + ln;
#pragma unroll
        for (int r = 0; r < 8; ++r) {
            int m = tileBase + mtile * 16 + h * 8 + r;
            out[(size_t)m * 256 + n] = xout[nt][r];
        }
    }
}

extern "C" void kernel_launch(void* const* d_in, const int* in_sizes, int n_in,
                              void* d_out, int out_size, void* d_ws, size_t ws_size,
                              hipStream_t stream) {
    const float* in_pos = (const float*)d_in[0];
    const float* table  = (const float*)d_in[1];
    const float* ffn_A  = (const float*)d_in[2];
    const float* W0     = (const float*)d_in[3];
    const float* b0     = (const float*)d_in[4];
    const float* Ws     = (const float*)d_in[5];
    const float* bs     = (const float*)d_in[6];
    const float* Wh     = (const float*)d_in[7];
    const float* bh     = (const float*)d_in[8];
    float* out = (float*)d_out;

    unsigned short* apad = (unsigned short*)d_ws;        // 4*256*32
    unsigned short* wst  = apad + 4 * 256 * 32;          // 4*256*256
    unsigned short* wht  = wst + 4 * 256 * 256;          // 4*256*256

    ffb_prep<<<1024, 256, 0, stream>>>(ffn_A, Ws, Wh, apad, wst, wht);

    size_t smem = (size_t)(64 * XSTR + 256 * XSTR + 64 * XSTR + 64 * 32) *
                      sizeof(unsigned short) +
                  192 * sizeof(float);                   // ~204 KB of 320 KB/WGP
    ffb_main<<<131072 / 64, 256, smem, stream>>>(in_pos, table, W0, b0, bs, bh,
                                                 apad, wst, wht, out);
}
